// QLORA_70463233458673
// MI455X (gfx1250) — matrix operand order repaired and checked by reading, not compiled, not run on hardware
//
#include <hip/hip_runtime.h>
#include <hip/hip_bf16.h>

// ---------------------------------------------------------------------------
// QLoRA fused GEMM for MI455X (gfx1250, wave32, WMMA)
//   Y = X @ (W_nf4 * c1 * c2)  +  (X @ L1) @ L2
// N=8192, D_IN=4096, D_OUT=4096, RANK=16, all fp32 in memory.
// Strategy: on-the-fly dequant -> bf16 -> v_wmma_f32_16x16x32_bf16,
// fp32 accumulate. LoRA folded in as one extra K-tile. Double-buffered LDS.
// This rev: packed bf16 conversion (v_cvt_pk) + 32-bit LDS stores.
// ---------------------------------------------------------------------------

#define D_IN   4096
#define D_OUT  4096
#define RANK   16
#define NROWS  8192

#define BM 128           // block tile M
#define BN 128           // block tile N
#define BK 32            // K step (= WMMA K for bf16)
#define LDT 40           // padded LDS row stride in bf16 elems (mult of 8 -> 16B aligned rows)
#define KT_TOTAL (D_IN / BK + 1)   // +1: iteration 0 is the LoRA (rank) tile

typedef __attribute__((ext_vector_type(16))) __bf16          v16bf;
typedef __attribute__((ext_vector_type(8)))  float           v8f;
typedef __attribute__((ext_vector_type(8)))  unsigned short  v8u;

union Frag {
    unsigned short s[16];
    v8u            h[2];
    v16bf          v;
};

// Pack two floats into two bf16 (RNE via native fptrunc -> v_cvt_pk_bf16_f32)
__device__ __forceinline__ unsigned pk2bf(float lo, float hi) {
    union { __bf16 b[2]; unsigned u; } p;
    p.b[0] = (__bf16)lo;
    p.b[1] = (__bf16)hi;
    return p.u;
}

// ---------------------------------------------------------------------------
// Kernel 1: XL1[m][r] = sum_k X[m][k] * L1[k][r]      (8192 x 16)
// One block per row; X row staged in LDS once (coalesced); split-K reduce.
// ---------------------------------------------------------------------------
__global__ __launch_bounds__(256) void xl1_kernel(const float* __restrict__ X,
                                                  const float* __restrict__ L1,
                                                  float* __restrict__ XL1) {
    __shared__ float sx[D_IN];
    __shared__ float part[16][16];   // [k-slice][rank]

    const int m   = blockIdx.x;
    const int tid = threadIdx.x;

    const float4* xrow = (const float4*)(X + (size_t)m * D_IN);
    #pragma unroll
    for (int j = 0; j < 4; ++j) {
        float4 v = xrow[tid + j * 256];
        float* d = &sx[(tid + j * 256) * 4];
        d[0] = v.x; d[1] = v.y; d[2] = v.z; d[3] = v.w;
    }
    __syncthreads();

    const int r = tid & 15;        // rank index
    const int s = tid >> 4;        // k slice (16 slices of 256)
    float acc = 0.0f;
    const int kbeg = s * (D_IN / 16);
    #pragma unroll 4
    for (int k = kbeg; k < kbeg + D_IN / 16; ++k)
        acc = __builtin_fmaf(sx[k], L1[(size_t)k * RANK + r], acc);

    part[s][r] = acc;
    __syncthreads();

    if (tid < 16) {
        float sum = 0.0f;
        #pragma unroll
        for (int i = 0; i < 16; ++i) sum += part[i][tid];
        XL1[(size_t)m * RANK + tid] = sum;
    }
}

// ---------------------------------------------------------------------------
// Stage one K-tile into LDS (bf16, packed 32-bit stores).
// kt==0 stages the LoRA tile ( A = [XL1 | 0], B = [L2 ; 0] );
// kt>=1 stages k0=(kt-1)*32 with fused dequant for W.
// B is stored transposed (n-major) so B fragments are contiguous b128 loads;
// each thread owns two adjacent k-rows so transposed stores pack (k,k+1).
// ---------------------------------------------------------------------------
__device__ __forceinline__ void stageTile(int kt, int gm, int gn,
                                          const float* __restrict__ X,
                                          const float* __restrict__ Wq,
                                          const float* __restrict__ C2,
                                          float c1,
                                          const float* __restrict__ XL1,
                                          const float* __restrict__ L2,
                                          unsigned short* sA,
                                          unsigned short* sB,
                                          int tid) {
    if (kt == 0) {
        // ---- LoRA tile: rank 16 padded with zeros to K=32 ----
        const int r  = tid >> 1;   // row (A) / col (B), 0..127
        const int hh = tid & 1;    // which 16-wide K half
        unsigned va[8], vb[8];
        if (hh == 0) {
            const float4* px = (const float4*)(XL1 + (size_t)(gm + r) * RANK);
            #pragma unroll
            for (int j = 0; j < 4; ++j) {
                float4 v = px[j];
                va[2*j+0] = pk2bf(v.x, v.y);
                va[2*j+1] = pk2bf(v.z, v.w);
            }
            #pragma unroll
            for (int j = 0; j < 8; ++j)
                vb[j] = pk2bf(L2[(size_t)(2*j)     * D_OUT + gn + r],
                              L2[(size_t)(2*j + 1) * D_OUT + gn + r]);
        } else {
            #pragma unroll
            for (int j = 0; j < 8; ++j) { va[j] = 0u; vb[j] = 0u; }
        }
        unsigned* dA = (unsigned*)(sA + r * LDT + hh * 16);
        unsigned* dB = (unsigned*)(sB + r * LDT + hh * 16);
        #pragma unroll
        for (int j = 0; j < 8; ++j) { dA[j] = va[j]; dB[j] = vb[j]; }
    } else {
        const int k0 = (kt - 1) * BK;
        // ---- A: X[gm..gm+127][k0..k0+31] -> bf16, row-major ----
        {
            const int r  = tid >> 1;
            const int hh = tid & 1;
            const float* base = X + (size_t)(gm + r) * D_IN + k0 + hh * 16;
            if (k0 + 2 * BK < D_IN)
                __builtin_prefetch(base + 2 * BK, 0, 1);   // global_prefetch_b8
            const float4* px = (const float4*)base;
            unsigned va[8];
            #pragma unroll
            for (int j = 0; j < 4; ++j) {
                float4 v = px[j];
                va[2*j+0] = pk2bf(v.x, v.y);
                va[2*j+1] = pk2bf(v.z, v.w);
            }
            unsigned* dA = (unsigned*)(sA + r * LDT + hh * 16);
            #pragma unroll
            for (int j = 0; j < 8; ++j) dA[j] = va[j];
        }
        // ---- B: dequant W[k0..k0+31][gn..gn+127] -> bf16, TRANSPOSED ----
        // Thread owns k rows (kk, kk+1) x 8 columns -> packed b32 stores.
        {
            const int kk = (tid & 15) * 2;    // 0,2,..,30
            const int n0 = (tid >> 4) * 8;    // 0..120 step 8
            const float* wb0 = Wq + (size_t)(k0 + kk)     * D_OUT + gn + n0;
            const float* wb1 = Wq + (size_t)(k0 + kk + 1) * D_OUT + gn + n0;
            const float* cb0 = C2 + (size_t)(k0 + kk)     * D_OUT + gn + n0;
            const float* cb1 = C2 + (size_t)(k0 + kk + 1) * D_OUT + gn + n0;
            if (k0 + 2 * BK < D_IN) {
                __builtin_prefetch(wb0 + (size_t)2 * BK * D_OUT, 0, 1);
                __builtin_prefetch(cb0 + (size_t)2 * BK * D_OUT, 0, 1);
            }
            float w0[8], w1[8];
            #pragma unroll
            for (int j = 0; j < 2; ++j) {
                float4 a0 = ((const float4*)wb0)[j];
                float4 a1 = ((const float4*)wb1)[j];
                float4 s0 = ((const float4*)cb0)[j];
                float4 s1 = ((const float4*)cb1)[j];
                w0[4*j+0] = a0.x * c1 * s0.x;  w1[4*j+0] = a1.x * c1 * s1.x;
                w0[4*j+1] = a0.y * c1 * s0.y;  w1[4*j+1] = a1.y * c1 * s1.y;
                w0[4*j+2] = a0.z * c1 * s0.z;  w1[4*j+2] = a1.z * c1 * s1.z;
                w0[4*j+3] = a0.w * c1 * s0.w;  w1[4*j+3] = a1.w * c1 * s1.w;
            }
            #pragma unroll
            for (int j = 0; j < 8; ++j)
                *(unsigned*)(sB + (n0 + j) * LDT + kk) = pk2bf(w0[j], w1[j]);
        }
    }
}

// ---------------------------------------------------------------------------
// Kernel 2: fused dequant GEMM + LoRA via WMMA.
// Block = 256 threads = 8 wave32s. Block tile 128x128; wave tile 64x32.
// ---------------------------------------------------------------------------
__global__ __launch_bounds__(256) void qlora_gemm(const float* __restrict__ X,
                                                  const float* __restrict__ Wq,
                                                  const float* __restrict__ c1p,
                                                  const float* __restrict__ C2,
                                                  const float* __restrict__ XL1,
                                                  const float* __restrict__ L2,
                                                  float* __restrict__ Y) {
    __shared__ alignas(16) unsigned short ldsA[2][BM * LDT];   // 2 x 10 KB
    __shared__ alignas(16) unsigned short ldsB[2][BN * LDT];   // 2 x 10 KB

    const int gm  = blockIdx.y * BM;
    const int gn  = blockIdx.x * BN;
    const int tid = threadIdx.x;
    const int lane = tid & 31;
    const int wv   = tid >> 5;            // 0..7
    const int wave_m = (wv >> 2) * 64;    // 0 or 64
    const int wave_n = (wv & 3) * 32;     // 0,32,64,96
    const int lr = lane & 15;
    const int hh = lane >> 4;

    const float c1 = c1p[0];

    v8f acc[4][2] = {};

    stageTile(0, gm, gn, X, Wq, C2, c1, XL1, L2, ldsA[0], ldsB[0], tid);
    __syncthreads();

    int p = 0;
    for (int kt = 0; kt < KT_TOTAL; ++kt) {
        if (kt + 1 < KT_TOTAL)
            stageTile(kt + 1, gm, gn, X, Wq, C2, c1, XL1, L2,
                      ldsA[p ^ 1], ldsB[p ^ 1], tid);

        // ---- WMMA compute on buffer p ----
        const unsigned short* sA = &ldsA[p][wave_m * LDT];
        const unsigned short* sB = &ldsB[p][wave_n * LDT];

        Frag a[4];
        #pragma unroll
        for (int mi = 0; mi < 4; ++mi) {
            // A 16x32 bf16 layout: lane lr holds row; 16 contiguous halves
            // at k = hh*16 (two 16B ds_load_b128)
            const unsigned short* pa = sA + (mi * 16 + lr) * LDT + hh * 16;
            a[mi].h[0] = *(const v8u*)pa;
            a[mi].h[1] = *(const v8u*)(pa + 8);
        }
        Frag b[2];
        #pragma unroll
        for (int ni = 0; ni < 2; ++ni) {
            // B 32x16 bf16 (stored n-major): lane lr holds column;
            // 16 contiguous halves at k = hh*16
            const unsigned short* pb = sB + (ni * 16 + lr) * LDT + hh * 16;
            b[ni].h[0] = *(const v8u*)pb;
            b[ni].h[1] = *(const v8u*)(pb + 8);
        }
        #pragma unroll
        for (int mi = 0; mi < 4; ++mi)
            #pragma unroll
            for (int ni = 0; ni < 2; ++ni)
                acc[mi][ni] = __builtin_amdgcn_wmma_f32_16x16x32_bf16(
                    false, a[mi].v, false, b[ni].v,
                    (short)0, acc[mi][ni], false, false);

        __syncthreads();
        p ^= 1;
    }

    // ---- Epilogue: C/D layout -> global. lane holds col lr, rows hh*8+i ----
    #pragma unroll
    for (int mi = 0; mi < 4; ++mi) {
        #pragma unroll
        for (int ni = 0; ni < 2; ++ni) {
            const int col = gn + wave_n + ni * 16 + lr;
            #pragma unroll
            for (int i = 0; i < 8; ++i) {
                const int row = gm + wave_m + mi * 16 + hh * 8 + i;
                Y[(size_t)row * D_OUT + col] = acc[mi][ni][i];
            }
        }
    }
}

// ---------------------------------------------------------------------------
// Host launch
// ---------------------------------------------------------------------------
extern "C" void kernel_launch(void* const* d_in, const int* in_sizes, int n_in,
                              void* d_out, int out_size, void* d_ws, size_t ws_size,
                              hipStream_t stream) {
    const float* X   = (const float*)d_in[0];   // [8192,4096]
    const float* Wq  = (const float*)d_in[1];   // [4096,4096]
    const float* c1  = (const float*)d_in[2];   // scalar
    const float* C2  = (const float*)d_in[3];   // [4096,4096]
    const float* L1  = (const float*)d_in[4];   // [4096,16]
    const float* L2  = (const float*)d_in[5];   // [16,4096]
    float*       Y   = (float*)d_out;           // [8192,4096]
    float*       XL1 = (float*)d_ws;            // [8192,16] scratch (512 KB)

    xl1_kernel<<<NROWS, 256, 0, stream>>>(X, L1, XL1);

    dim3 grid(D_OUT / BN, NROWS / BM);          // (32, 64)
    qlora_gemm<<<grid, 256, 0, stream>>>(X, Wq, c1, C2, XL1, L2, Y);
}
